// DisentangledSelfAttention_62603443306699
// MI455X (gfx1250) — compile-verified, matrix operand order checked by reading, not run on hardware
//
#include <hip/hip_runtime.h>
#include <hip/hip_bf16.h>
#include <math.h>

typedef __bf16 bf16;
typedef __attribute__((ext_vector_type(16))) __bf16 v16bf;
typedef __attribute__((ext_vector_type(8)))  float  v8f;
typedef unsigned int u32x4 __attribute__((ext_vector_type(4)));
typedef int          i32x8 __attribute__((ext_vector_type(8)));
typedef int          i32x4 __attribute__((ext_vector_type(4)));

constexpr int kH   = 1024;   // hidden
constexpr int kNH  = 16;     // heads
constexpr int kHD  = 64;     // head dim
constexpr int kS   = 2048;   // seq len
constexpr int kB   = 2;      // batch
constexpr int kM   = kB * kS;    // 4096 rows
constexpr int kP2  = 512;        // 2*SPAN
constexpr int kSPAN = 256;

// LDS byte offsets inside the attention kernel's dynamic LDS block (176 KB total)
constexpr unsigned kOffQ    = 0;        //  64*64 bf16 =   8 KB
constexpr unsigned kOffC2P  = 8192;     //  64*512 f32 = 128 KB
constexpr unsigned kOffK0   = 139264;   //  64*64 bf16 =   8 KB (double-buffered)
constexpr unsigned kOffK1   = 147456;
constexpr unsigned kOffVt0  = 155648;   //  64*64 bf16 =   8 KB (double-buffered)
constexpr unsigned kOffVt1  = 163840;
constexpr unsigned kOffP    = 172032;   // 4*16*64 bf16 =  8 KB
constexpr unsigned kLdsBytes = 180224;

__device__ __forceinline__ v8f wmma_bf16(v16bf a, v16bf b, v8f c) {
  return __builtin_amdgcn_wmma_f32_16x16x32_bf16(false, a, false, b, (short)0, c, false, false);
}

// TDM: async-load a 64x64 bf16 tile (row pitch = stride_elems) from global to LDS.
// D# bit packing per CDNA5 ISA 8.3/8.4 (group0: count/lds/global/type, group1: dims).
__device__ __forceinline__ void tdm_load_tile64(unsigned lds_off, const bf16* gptr,
                                                unsigned stride_elems) {
  unsigned long long ga = (unsigned long long)(size_t)gptr;
  u32x4 g0 = { 1u,                                             // count=1 (valid, user)
               lds_off,                                        // lds_addr [63:32]
               (unsigned)(ga & 0xffffffffu),                   // global_addr [95:64]
               (unsigned)((ga >> 32) & 0x1ffffffu) | (2u << 30) }; // addr[56:32] | type=2
  i32x8 g1 = { (int)(1u << 16),        // [17:16] data_size=1 (2 bytes); mask/flags=0
               (int)(64u << 16),       // [63:48]  tensor_dim0.lo = 64
               (int)(64u << 16),       // [79:64]  tdim0.hi=0 | [95:80] tensor_dim1.lo=64
               (int)(64u << 16),       // [111:96] tdim1.hi=0 | [127:112] tile_dim0=64
               64,                     // [143:128] tile_dim1=64 | [159:144] tile_dim2=0
               (int)stride_elems,      // [191:160] tensor_dim0_stride.lo
               0, 0 };                 // stride0.hi=0, tensor_dim1_stride=0
  i32x4 z4 = { 0, 0, 0, 0 };
  i32x8 z8 = { 0, 0, 0, 0, 0, 0, 0, 0 };
  __builtin_amdgcn_tensor_load_to_lds(g0, g1, z4, z4, z8, 0);
}

// ---------------- fp32 -> bf16 convert ----------------
__global__ void cvt_f32_bf16(const float* __restrict__ src, bf16* __restrict__ dst, int n) {
  for (int i = blockIdx.x * blockDim.x + threadIdx.x; i < n; i += gridDim.x * blockDim.x)
    dst[i] = (bf16)src[i];
}

// ---------------- fp32 [K,N] -> bf16 transposed [N,K] ----------------
__global__ void cvt_transpose(const float* __restrict__ src, bf16* __restrict__ dst,
                              int K, int N) {
  for (int i = blockIdx.x * blockDim.x + threadIdx.x; i < K * N; i += gridDim.x * blockDim.x) {
    int k = i / N, n = i % N;
    dst[(size_t)n * K + k] = (bf16)src[i];
  }
}

// ---------------- LayerNorm of rel_emb rows -> bf16 ----------------
__global__ void ln_kernel(const float* __restrict__ re, const float* __restrict__ g,
                          const float* __restrict__ be, bf16* __restrict__ out) {
  __shared__ float red[256];
  int row = blockIdx.x, t = threadIdx.x;
  const float* x = re + (size_t)row * kH;
  float s = 0.f;
  for (int c = t; c < kH; c += 256) s += x[c];
  red[t] = s; __syncthreads();
  for (int st = 128; st > 0; st >>= 1) { if (t < st) red[t] += red[t + st]; __syncthreads(); }
  float mu = red[0] * (1.f / kH);
  __syncthreads();
  float v = 0.f;
  for (int c = t; c < kH; c += 256) { float d = x[c] - mu; v += d * d; }
  red[t] = v; __syncthreads();
  for (int st = 128; st > 0; st >>= 1) { if (t < st) red[t] += red[t + st]; __syncthreads(); }
  float rstd = rsqrtf(red[0] * (1.f / kH) + 1e-5f);
  for (int c = t; c < kH; c += 256)
    out[(size_t)row * kH + c] = (bf16)((x[c] - mu) * rstd * g[c] + be[c]);
}

// ---------------- log-bucket index tables over delta = q-k ----------------
__global__ void bucket_kernel(int* __restrict__ c2pIdx, int* __restrict__ p2cIdx) {
  int i = blockIdx.x * blockDim.x + threadIdx.x;
  if (i >= 2 * kS - 1) return;
  int d = i - (kS - 1);
  const int mid = 128;                       // BUCKETS/2
  float sgn = (d > 0) ? 1.f : ((d < 0) ? -1.f : 0.f);
  float abs_pos = (d < mid && d > -mid) ? (float)(mid - 1) : fabsf((float)d);
  float log_pos = ceilf(logf(abs_pos / (float)mid) /
                        logf((float)(512 - 1) / (float)mid) * (float)(mid - 1)) + (float)mid;
  float bucket = (abs_pos <= (float)mid) ? (float)d : log_pos * sgn;
  int bi = (int)bucket;
  int a = bi + kSPAN;  a = a < 0 ? 0 : (a > kP2 - 1 ? kP2 - 1 : a);
  int b = -bi + kSPAN; b = b < 0 ? 0 : (b > kP2 - 1 ? kP2 - 1 : b);
  c2pIdx[i] = a;
  p2cIdx[i] = b;
}

// ---------------- tiled bf16 WMMA GEMM: C[M,N] = A[M,K] @ Wt[N,K]^T + bias ----------------
// Wt is PRE-TRANSPOSED [N][K] so both LDS stages are pure b128 copies.
// grid (N/64, M/64), block 128 (4 waves); each wave owns a 16x64 stripe.
// VTOUT: scatter the output as V-transposed [B][NH][HD][S] (for attention PV B-matrix).
template <bool F32OUT, bool VTOUT>
__global__ void gemm_kernel(const bf16* __restrict__ A, const bf16* __restrict__ Wt,
                            const float* __restrict__ bias, void* __restrict__ Cout,
                            int M, int N, int K) {
  __shared__ bf16 sA[64 * 32];
  __shared__ bf16 sBt[64 * 32];   // [n][k]
  int n0 = blockIdx.x * 64, m0 = blockIdx.y * 64;
  int tid = threadIdx.x, w = tid >> 5, l = tid & 31, ln = l & 15, kh = l >> 4;
  v8f acc[4] = {};
  for (int k0 = 0; k0 < K; k0 += 32) {
    { // A tile 64x32
      int r = tid >> 1, c0 = (tid & 1) * 16;
      *(v16bf*)&sA[r * 32 + c0] = *(const v16bf*)&A[(size_t)(m0 + r) * K + k0 + c0];
    }
    { // Wt tile 64x32 (already [n][k] in memory)
      int n = tid >> 1, c0 = (tid & 1) * 16;
      *(v16bf*)&sBt[n * 32 + c0] = *(const v16bf*)&Wt[(size_t)(n0 + n) * K + k0 + c0];
    }
    __syncthreads();
    v16bf af = *(const v16bf*)&sA[(w * 16 + ln) * 32 + kh * 16];
    for (int ns = 0; ns < 4; ++ns) {
      v16bf bfrag = *(const v16bf*)&sBt[(ns * 16 + ln) * 32 + kh * 16];
      acc[ns] = wmma_bf16(af, bfrag, acc[ns]);
    }
    __syncthreads();
  }
  for (int ns = 0; ns < 4; ++ns)
    for (int i = 0; i < 8; ++i) {
      int row = m0 + w * 16 + i + 8 * kh;
      int col = n0 + ns * 16 + ln;
      float v = acc[ns][i] + bias[col];
      if (F32OUT) {
        ((float*)Cout)[(size_t)row * N + col] = v;
      } else if (VTOUT) {
        int bb = row / kS, ss = row % kS, hh = col >> 6, dd = col & 63;
        ((bf16*)Cout)[(((size_t)(bb * kNH + hh)) * kHD + dd) * kS + ss] = (bf16)v;
      } else {
        ((bf16*)Cout)[(size_t)row * N + col] = (bf16)v;
      }
    }
}

// ---------------- p2c_all[h][m][p] = K_h[m,:] . pos_q_h[p,:]  (K=64) ----------------
// grid (512/64, 4096/64, NH), block 128
__global__ void p2c_kernel(const bf16* __restrict__ Kmat, const bf16* __restrict__ posq,
                           float* __restrict__ out) {
  __shared__ bf16 sA[64 * 64];
  __shared__ bf16 sB[64 * 64];
  int p0 = blockIdx.x * 64, m0 = blockIdx.y * 64, h = blockIdx.z;
  int tid = threadIdx.x, w = tid >> 5, l = tid & 31, ln = l & 15, kh = l >> 4;
  {
    int r = tid >> 1, c0 = (tid & 1) * 32;
    *(v16bf*)&sA[r * 64 + c0]      = *(const v16bf*)&Kmat[(size_t)(m0 + r) * kH + h * 64 + c0];
    *(v16bf*)&sA[r * 64 + c0 + 16] = *(const v16bf*)&Kmat[(size_t)(m0 + r) * kH + h * 64 + c0 + 16];
    *(v16bf*)&sB[r * 64 + c0]      = *(const v16bf*)&posq[(size_t)(p0 + r) * kH + h * 64 + c0];
    *(v16bf*)&sB[r * 64 + c0 + 16] = *(const v16bf*)&posq[(size_t)(p0 + r) * kH + h * 64 + c0 + 16];
  }
  __syncthreads();
  v8f acc[4] = {};
  for (int ks = 0; ks < 2; ++ks) {
    v16bf af = *(const v16bf*)&sA[(w * 16 + ln) * 64 + ks * 32 + kh * 16];
    for (int ns = 0; ns < 4; ++ns) {
      v16bf bfrag = *(const v16bf*)&sB[(ns * 16 + ln) * 64 + ks * 32 + kh * 16];
      acc[ns] = wmma_bf16(af, bfrag, acc[ns]);
    }
  }
  for (int ns = 0; ns < 4; ++ns)
    for (int i = 0; i < 8; ++i)
      out[((size_t)h * kM + m0 + w * 16 + i + 8 * kh) * kP2 + p0 + ns * 16 + ln] = acc[ns][i];
}

// ---------------- flash attention with disentangled biases ----------------
// grid (S/64, NH, B), block 128 (4 waves). Dynamic LDS: 176 KB.
// K/Vt tiles are double-buffered and fetched by the Tensor Data Mover so the DMA of
// tile kt+1 overlaps the WMMA + softmax work on tile kt.
__global__ void attn_kernel(const bf16* __restrict__ Qb, const bf16* __restrict__ Kb,
                            const bf16* __restrict__ Vtb, const bf16* __restrict__ poskb,
                            const float* __restrict__ p2c_all,
                            const int* __restrict__ c2pIdx, const int* __restrict__ p2cIdx,
                            bf16* __restrict__ ctx) {
  extern __shared__ char smem[];
  bf16*  sQ   = (bf16*)(smem + kOffQ);
  float* sC2P = (float*)(smem + kOffC2P);
  bf16*  sP   = (bf16*)(smem + kOffP);
  const unsigned offK[2]  = { kOffK0,  kOffK1 };
  const unsigned offVt[2] = { kOffVt0, kOffVt1 };
  const float INV_SCALE = 0.07216878364870323f;  // 1/sqrt(64*3)

  int qt = blockIdx.x, h = blockIdx.y, b = blockIdx.z;
  int q0 = qt * 64;
  int tid = threadIdx.x, w = tid >> 5, l = tid & 31, ln = l & 15, kh = l >> 4;

  const bf16* Kbase  = Kb  + ((size_t)(b * kS)) * kH + h * 64;             // row pitch kH
  const bf16* Vtbase = Vtb + ((size_t)((b * kNH + h)) * kHD) * kS;         // row pitch kS

  { // load Q tile 64x64
    int r = tid >> 1, c0 = (tid & 1) * 32;
    const bf16* src = Qb + ((size_t)(b * kS + q0 + r)) * kH + h * 64 + c0;
    *(v16bf*)&sQ[r * 64 + c0]      = *(const v16bf*)src;
    *(v16bf*)&sQ[r * 64 + c0 + 16] = *(const v16bf*)(src + 16);
  }
  __syncthreads();
  v16bf aQ0 = *(const v16bf*)&sQ[(w * 16 + ln) * 64 + kh * 16];
  v16bf aQ1 = *(const v16bf*)&sQ[(w * 16 + ln) * 64 + 32 + kh * 16];

  // c2p = q_tile @ pos_k^T  ->  sC2P[64][512]   (stages pos_k through buffer 0's region)
  bf16* sPos = (bf16*)(smem + kOffK0);
  for (int pt = 0; pt < 8; ++pt) {
    { int r = tid >> 1, c0 = (tid & 1) * 32;
      const bf16* src = poskb + ((size_t)(pt * 64 + r)) * kH + h * 64 + c0;
      *(v16bf*)&sPos[r * 64 + c0]      = *(const v16bf*)src;
      *(v16bf*)&sPos[r * 64 + c0 + 16] = *(const v16bf*)(src + 16);
    }
    __syncthreads();
    v8f acc[4] = {};
    for (int ns = 0; ns < 4; ++ns) {
      v16bf b0 = *(const v16bf*)&sPos[(ns * 16 + ln) * 64 + kh * 16];
      v16bf b1 = *(const v16bf*)&sPos[(ns * 16 + ln) * 64 + 32 + kh * 16];
      acc[ns] = wmma_bf16(aQ0, b0, acc[ns]);
      acc[ns] = wmma_bf16(aQ1, b1, acc[ns]);
    }
    for (int ns = 0; ns < 4; ++ns)
      for (int i = 0; i < 8; ++i)
        sC2P[(w * 16 + i + 8 * kh) * kP2 + pt * 64 + ns * 16 + ln] = acc[ns][i];
    __syncthreads();
  }

  // pipeline prologue: DMA tile 0 into buffer 0
  if (tid < 32) {
    tdm_load_tile64(offK[0],  Kbase,  kH);
    tdm_load_tile64(offVt[0], Vtbase, kS);
  }

  float mrow[8], lrow[8];
  for (int i = 0; i < 8; ++i) { mrow[i] = -1e30f; lrow[i] = 0.f; }
  v8f o[4] = {};

  for (int kt = 0; kt < kS / 64; ++kt) {
    int cur = kt & 1;
    if (tid < 32) __builtin_amdgcn_s_wait_tensorcnt(0);   // tile kt resident
    __syncthreads();   // publishes buf[cur]; retires iter kt-1 reads of buf[cur^1]
    if (kt + 1 < kS / 64 && tid < 32) {                   // prefetch tile kt+1 (overlapped)
      tdm_load_tile64(offK[cur ^ 1],  Kbase  + (size_t)(kt + 1) * 64 * kH, kH);
      tdm_load_tile64(offVt[cur ^ 1], Vtbase + (kt + 1) * 64, kS);
    }
    const bf16* sK  = (const bf16*)(smem + offK[cur]);
    const bf16* sVt = (const bf16*)(smem + offVt[cur]);   // [d][key]

    v8f s[4] = {};
    for (int ns = 0; ns < 4; ++ns) {
      v16bf b0 = *(const v16bf*)&sK[(ns * 16 + ln) * 64 + kh * 16];
      v16bf b1 = *(const v16bf*)&sK[(ns * 16 + ln) * 64 + 32 + kh * 16];
      s[ns] = wmma_bf16(aQ0, b0, s[ns]);
      s[ns] = wmma_bf16(aQ1, b1, s[ns]);
    }

    // add disentangled biases, uniform scale
    float arr[4][8];
    for (int ns = 0; ns < 4; ++ns) {
      int kg = kt * 64 + ns * 16 + ln;
      const float* p2crow = p2c_all + ((size_t)h * kM + (size_t)b * kS + kg) * kP2;
      for (int i = 0; i < 8; ++i) {
        int ql = w * 16 + i + 8 * kh;               // local q row in tile
        int t = (q0 + ql) - kg + (kS - 1);
        float v = s[ns][i] + sC2P[ql * kP2 + c2pIdx[t]] + p2crow[p2cIdx[t]];
        arr[ns][i] = v * INV_SCALE;
      }
    }

    // online softmax (rows live in 16-lane halves; xor masks 1..8 stay in-half)
    for (int i = 0; i < 8; ++i) {
      float tm = fmaxf(fmaxf(arr[0][i], arr[1][i]), fmaxf(arr[2][i], arr[3][i]));
      for (int off = 8; off >= 1; off >>= 1) tm = fmaxf(tm, __shfl_xor(tm, off, 32));
      float mnew = fmaxf(mrow[i], tm);
      float corr = __expf(mrow[i] - mnew);
      mrow[i] = mnew;
      float ps = 0.f;
      for (int ns = 0; ns < 4; ++ns) { arr[ns][i] = __expf(arr[ns][i] - mnew); ps += arr[ns][i]; }
      for (int off = 8; off >= 1; off >>= 1) ps += __shfl_xor(ps, off, 32);
      lrow[i] = lrow[i] * corr + ps;
      for (int ns = 0; ns < 4; ++ns) o[ns][i] *= corr;
    }

    // P: C-layout -> A-layout through LDS, then O += P @ V
    bf16* sPw = sP + w * (16 * 64);
    for (int ns = 0; ns < 4; ++ns)
      for (int i = 0; i < 8; ++i)
        sPw[(i + 8 * kh) * 64 + ns * 16 + ln] = (bf16)arr[ns][i];
    __syncthreads();
    v16bf aP0 = *(const v16bf*)&sPw[ln * 64 + kh * 16];
    v16bf aP1 = *(const v16bf*)&sPw[ln * 64 + 32 + kh * 16];
    for (int ns = 0; ns < 4; ++ns) {
      v16bf b0 = *(const v16bf*)&sVt[(ns * 16 + ln) * 64 + kh * 16];
      v16bf b1 = *(const v16bf*)&sVt[(ns * 16 + ln) * 64 + 32 + kh * 16];
      o[ns] = wmma_bf16(aP0, b0, o[ns]);
      o[ns] = wmma_bf16(aP1, b1, o[ns]);
    }
    // no trailing barrier needed: next iteration's top barrier fences buf reuse
  }

  for (int ns = 0; ns < 4; ++ns)
    for (int i = 0; i < 8; ++i) {
      int qg = q0 + w * 16 + i + 8 * kh;
      float val = o[ns][i] / lrow[i];
      ctx[((size_t)(b * kS + qg)) * kH + h * 64 + ns * 16 + ln] = (bf16)val;
    }
}

extern "C" void kernel_launch(void* const* d_in, const int* in_sizes, int n_in,
                              void* d_out, int out_size, void* d_ws, size_t ws_size,
                              hipStream_t stream) {
  (void)in_sizes; (void)n_in; (void)out_size; (void)ws_size;
  const float* hs  = (const float*)d_in[0];
  const float* Wq  = (const float*)d_in[1];  const float* bq  = (const float*)d_in[2];
  const float* Wk  = (const float*)d_in[3];  const float* bk  = (const float*)d_in[4];
  const float* Wv  = (const float*)d_in[5];  const float* bv  = (const float*)d_in[6];
  const float* Wo  = (const float*)d_in[7];  const float* bo  = (const float*)d_in[8];
  const float* Wpk = (const float*)d_in[9];  const float* bpk = (const float*)d_in[10];
  const float* Wpq = (const float*)d_in[11]; const float* bpq = (const float*)d_in[12];
  const float* rel = (const float*)d_in[13];
  const float* lng = (const float*)d_in[14]; const float* lnb = (const float*)d_in[15];

  char* ws = (char*)d_ws;
  size_t off = 0;
  auto carve = [&](size_t bytes) -> char* {
    char* p = ws + off;
    off = (off + bytes + 255) & ~(size_t)255;
    return p;
  };
  bf16* Xb    = (bf16*)carve((size_t)kM * kH * 2);
  bf16* Wqt   = (bf16*)carve((size_t)kH * kH * 2);   // transposed [N][K]
  bf16* Wkt   = (bf16*)carve((size_t)kH * kH * 2);
  bf16* Wvt   = (bf16*)carve((size_t)kH * kH * 2);
  bf16* Wot   = (bf16*)carve((size_t)kH * kH * 2);
  bf16* Wpkt  = (bf16*)carve((size_t)kH * kH * 2);
  bf16* Wpqt  = (bf16*)carve((size_t)kH * kH * 2);
  bf16* reb   = (bf16*)carve((size_t)kP2 * kH * 2);
  bf16* Qb    = (bf16*)carve((size_t)kM * kH * 2);
  bf16* Kb    = (bf16*)carve((size_t)kM * kH * 2);
  bf16* Vtb   = (bf16*)carve((size_t)kM * kH * 2);   // [B][NH][HD][S]
  bf16* poskb = (bf16*)carve((size_t)kP2 * kH * 2);
  bf16* posqb = (bf16*)carve((size_t)kP2 * kH * 2);
  bf16* ctxb  = (bf16*)carve((size_t)kM * kH * 2);
  int*  c2pI  = (int*)carve((size_t)(2 * kS - 1) * 4);
  int*  p2cI  = (int*)carve((size_t)(2 * kS - 1) * 4);
  float* p2cA = (float*)carve((size_t)kNH * kM * kP2 * 4);   // 128 MB (L2-resident)

  dim3 blk(128);
  cvt_f32_bf16<<<1024, 256, 0, stream>>>(hs, Xb, kM * kH);
  cvt_transpose<<<512, 256, 0, stream>>>(Wq,  Wqt,  kH, kH);
  cvt_transpose<<<512, 256, 0, stream>>>(Wk,  Wkt,  kH, kH);
  cvt_transpose<<<512, 256, 0, stream>>>(Wv,  Wvt,  kH, kH);
  cvt_transpose<<<512, 256, 0, stream>>>(Wo,  Wot,  kH, kH);
  cvt_transpose<<<512, 256, 0, stream>>>(Wpk, Wpkt, kH, kH);
  cvt_transpose<<<512, 256, 0, stream>>>(Wpq, Wpqt, kH, kH);
  ln_kernel<<<kP2, 256, 0, stream>>>(rel, lng, lnb, reb);
  bucket_kernel<<<16, 256, 0, stream>>>(c2pI, p2cI);

  // projections (V written pre-transposed for the attention PV B-matrix)
  gemm_kernel<false, false><<<dim3(kH / 64, kM / 64), blk, 0, stream>>>(Xb, Wqt, bq, Qb, kM, kH, kH);
  gemm_kernel<false, false><<<dim3(kH / 64, kM / 64), blk, 0, stream>>>(Xb, Wkt, bk, Kb, kM, kH, kH);
  gemm_kernel<false, true ><<<dim3(kH / 64, kM / 64), blk, 0, stream>>>(Xb, Wvt, bv, Vtb, kM, kH, kH);
  gemm_kernel<false, false><<<dim3(kH / 64, kP2 / 64), blk, 0, stream>>>(reb, Wpkt, bpk, poskb, kP2, kH, kH);
  gemm_kernel<false, false><<<dim3(kH / 64, kP2 / 64), blk, 0, stream>>>(reb, Wpqt, bpq, posqb, kP2, kH, kH);

  // p2c bias precompute
  p2c_kernel<<<dim3(kP2 / 64, kM / 64, kNH), blk, 0, stream>>>(Kb, posqb, p2cA);

  // flash attention with disentangled biases (double-buffered TDM-fed K/V tiles)
  attn_kernel<<<dim3(kS / 64, kNH, kB), blk, kLdsBytes, stream>>>(
      Qb, Kb, Vtb, poskb, p2cA, c2pI, p2cI, ctxb);

  // output projection -> fp32 d_out
  gemm_kernel<true, false><<<dim3(kH / 64, kM / 64), blk, 0, stream>>>(ctxb, Wot, bo, d_out, kM, kH, kH);
}
